// selfAttnLayer_20667382628450
// MI455X (gfx1250) — matrix operand order repaired and checked by reading, not compiled, not run on hardware
//
#include <hip/hip_runtime.h>
#include <hip/hip_bf16.h>

typedef __attribute__((ext_vector_type(16))) _Float16 v16h;
typedef __attribute__((ext_vector_type(8)))  _Float16 v8h;
typedef __attribute__((ext_vector_type(8)))  float    v8f;
typedef __attribute__((ext_vector_type(4)))  int      v4i;

#define N_NODES 50000
#define IC 256          // IN_CH
#define P  1024         // OUT_CH * N_HEADS (projection width)
#define D  256          // OUT_CH (= value dim)
#define NH 4
#define LDSK 40         // LDS row stride in halves (80B: b128-aligned, bank-spread)

// workspace layout (float offsets from start of d_ws)
#define SSQ_F   0                      // [0]=sum(q^2), [1]=sum(k^2)
#define KSUM_F  16                     // 1024 floats
#define KV_F    2048                   // 1024*256 floats
#define ZERO_TOTAL_F (KV_F + P * D)    // 264192 floats to clear
#define QOFF_B  ((size_t)(2u * 1024u * 1024u))                 // Q as f16, N x 1024
#define KOFF_B  (QOFF_B + (size_t)N_NODES * (size_t)P * 2u)    // K as f16, N x 1024

// ---------------- CDNA5 async global->LDS (guarded) ----------------

#if __has_builtin(__builtin_amdgcn_global_load_async_to_lds_b128)
#define HAVE_ASYNC_LDS 1
typedef __attribute__((address_space(1))) v4i g_v4i;   // global (AS1)
typedef __attribute__((address_space(3))) v4i l_v4i;   // LDS (AS3)
__device__ inline void async_ld_b128(const _Float16* g, _Float16* l) {
    __builtin_amdgcn_global_load_async_to_lds_b128((g_v4i*)g, (l_v4i*)l, 0, 0);
}
__device__ inline void wait_async0() {
#if __has_builtin(__builtin_amdgcn_s_wait_asynccnt)
    __builtin_amdgcn_s_wait_asynccnt(0);
#else
    asm volatile("s_wait_asynccnt 0x0" ::: "memory");
#endif
}
#endif

// ---------------- helpers ----------------

__device__ inline v8f zero8f() {
    v8f v;
#pragma unroll
    for (int i = 0; i < 8; ++i) v[i] = 0.0f;
    return v;
}

__device__ inline void st4h(_Float16* d, float4 v) {
    d[0] = (_Float16)v.x; d[1] = (_Float16)v.y;
    d[2] = (_Float16)v.z; d[3] = (_Float16)v.w;
}

// A fragment: 16x32 f16, row-major tile in LDS with stride LDSK halves. (ISA 7.12.2)
__device__ inline v16h load_fragA(const _Float16* tile, int rowBase) {
    const int lane = threadIdx.x & 31;
    const _Float16* p = tile + (size_t)(rowBase + (lane & 15)) * LDSK;
    const int hi = (lane >> 4) << 3;           // 0 or 8
    v8h lo = *(const v8h*)(p + hi);            // K = hi .. hi+7
    v8h hh = *(const v8h*)(p + 16 + hi);       // K = 16+hi .. 16+hi+7
    v16h r;
#pragma unroll
    for (int i = 0; i < 8; ++i) { r[i] = lo[i]; r[i + 8] = hh[i]; }
    return r;
}

// B fragment: 32x16 f16 stored transposed in LDS as F[n][k]; lane = N column,
// lanes 0-15 hold K=0..7,16..23, lanes 16-31 hold K=8..15,24..31 (same as A).
__device__ inline v16h load_fragB(const _Float16* tile, int rowBase) {
    const int lane = threadIdx.x & 31;
    const _Float16* p = tile + (size_t)(rowBase + (lane & 15)) * LDSK + ((lane >> 4) << 3);
    v8h lo = *(const v8h*)(p);
    v8h hh = *(const v8h*)(p + 16);
    v16h r;
#pragma unroll
    for (int i = 0; i < 8; ++i) { r[i] = lo[i]; r[i + 8] = hh[i]; }
    return r;
}

__device__ inline v8f wmma32(v16h a, v16h b, v8f c) {
    return __builtin_amdgcn_wmma_f32_16x16x32_f16(
        false, a, false, b, (short)0, c, false, false);
}

// ---------------- kernel 0: zero accumulators ----------------

__global__ void __launch_bounds__(256) zero_kernel(float* __restrict__ p, int n) {
    int i = (blockIdx.x * 256 + threadIdx.x) * 4;
    if (i < n) *(float4*)(p + i) = make_float4(0.f, 0.f, 0.f, 0.f);
}

// ---------------- kernel 1: projections (z=0: Q, z=1: K) + sum-of-squares ----

__global__ void __launch_bounds__(256) proj_kernel(
    const float* __restrict__ q_in, const float* __restrict__ s_in,
    const float* __restrict__ wq, const float* __restrict__ bq,
    const float* __restrict__ wk, const float* __restrict__ bk,
    float* __restrict__ ssq, _Float16* __restrict__ qout, _Float16* __restrict__ kout)
{
    const int z = blockIdx.z;
    const float* X  = z ? s_in : q_in;
    const float* W  = z ? wk : wq;
    const float* Bv = z ? bk : bq;
    _Float16* Out   = z ? kout : qout;

    __shared__ _Float16 lA[2][128 * LDSK];
    __shared__ _Float16 lB[2][128 * LDSK];
    __shared__ float red[256];

    const int tid  = threadIdx.x;
    const int lane = tid & 31;
    const int wid  = tid >> 5;
    const int wm   = (wid & 3) * 32;
    const int wn   = (wid >> 2) * 64;
    const int mBase = blockIdx.x * 128;
    const int nBase = blockIdx.y * 128;

    // precomputed staging addresses (clamped row; garbage rows are never stored)
    const float* pA[4];
    const float* pB[4];
    int offL[4];
#pragma unroll
    for (int i = 0; i < 4; ++i) {
        int idx = tid + i * 256;            // float4 id within 128x8
        int row = idx >> 3, c4 = (idx & 7) * 4;
        int gr  = mBase + row; if (gr >= N_NODES) gr = N_NODES - 1;
        pA[i]   = X + (size_t)gr * IC + c4;
        pB[i]   = W + (size_t)(nBase + row) * IC + c4;
        offL[i] = row * LDSK + c4;
    }

    v8f acc[2][4];
#pragma unroll
    for (int ti = 0; ti < 2; ++ti)
#pragma unroll
        for (int tj = 0; tj < 4; ++tj) acc[ti][tj] = zero8f();

    // prologue: stage chunk 0
    float4 ra[4], rb[4];
#pragma unroll
    for (int i = 0; i < 4; ++i) { ra[i] = *(const float4*)pA[i]; rb[i] = *(const float4*)pB[i]; }
#pragma unroll
    for (int i = 0; i < 4; ++i) { st4h(&lA[0][offL[i]], ra[i]); st4h(&lB[0][offL[i]], rb[i]); }

#pragma unroll 1
    for (int kc = 0; kc < IC / 32; ++kc) {
        __syncthreads();
        const int cur = kc & 1;
        const bool more = (kc + 1) < IC / 32;
        if (more) {
#pragma unroll
            for (int i = 0; i < 4; ++i) {
                ra[i] = *(const float4*)(pA[i] + (kc + 1) * 32);
                rb[i] = *(const float4*)(pB[i] + (kc + 1) * 32);
            }
        }
        v16h af[2], bf[4];
#pragma unroll
        for (int ti = 0; ti < 2; ++ti) af[ti] = load_fragA(lA[cur], wm + ti * 16);
#pragma unroll
        for (int tj = 0; tj < 4; ++tj) bf[tj] = load_fragB(lB[cur], wn + tj * 16);
#pragma unroll
        for (int ti = 0; ti < 2; ++ti)
#pragma unroll
            for (int tj = 0; tj < 4; ++tj)
                acc[ti][tj] = wmma32(af[ti], bf[tj], acc[ti][tj]);
        if (more) {
#pragma unroll
            for (int i = 0; i < 4; ++i) {
                st4h(&lA[cur ^ 1][offL[i]], ra[i]);
                st4h(&lB[cur ^ 1][offL[i]], rb[i]);
            }
        }
    }

    // epilogue: + bias, f16 store, fused sum-of-squares (valid rows only)
    float ssq_loc = 0.f;
    const int cn  = lane & 15;
    const int cm0 = (lane >> 4) * 8;
#pragma unroll
    for (int tj = 0; tj < 4; ++tj) {
        int nG = nBase + wn + tj * 16 + cn;
        float bias = Bv[nG];
#pragma unroll
        for (int ti = 0; ti < 2; ++ti) {
#pragma unroll
            for (int i = 0; i < 8; ++i) {
                int mG = mBase + wm + ti * 16 + cm0 + i;
                if (mG < N_NODES) {
                    float val = acc[ti][tj][i] + bias;
                    Out[(size_t)mG * P + nG] = (_Float16)val;
                    ssq_loc += val * val;
                }
            }
        }
    }
    red[tid] = ssq_loc;
    __syncthreads();
#pragma unroll
    for (int s = 128; s > 0; s >>= 1) {
        if (tid < s) red[tid] += red[tid + s];
        __syncthreads();
    }
    if (tid == 0) atomicAdd(&ssq[z], red[0]);
}

// ---------------- kernel 2: kv = K^T V (split-K over nodes) + ksum ----------
// Nodes are the reduction axis here, so out-of-range lanes MUST be zeroed.

__global__ void __launch_bounds__(256) kv_kernel(
    const float* __restrict__ src, const _Float16* __restrict__ kf,
    float* __restrict__ kv, float* __restrict__ ksum)
{
    __shared__ _Float16 lA[2][128 * LDSK];  // K^T tile: [chan 128][node 32]
    __shared__ _Float16 lB[2][128 * LDSK];  // V^T tile: [d 128][node 32]

    const int tid  = threadIdx.x;
    const int lane = tid & 31;
    const int wid  = tid >> 5;
    const int wm   = (wid & 3) * 32;
    const int wn   = (wid >> 2) * 64;
    const int mBase = blockIdx.x * 128;  // channel tile (of 1024)
    const int dBase = blockIdx.y * 128;  // d tile (of 256)
    const int chunk  = (N_NODES + gridDim.z - 1) / gridDim.z;
    const int nStart = blockIdx.z * chunk;
    const int nEnd   = min(nStart + chunk, N_NODES);
    const int nIter  = (nEnd - nStart + 31) / 32;

    const int nl  = tid >> 3;   // node local 0..31
    const int seg = tid & 7;

    v8f acc[2][4];
#pragma unroll
    for (int ti = 0; ti < 2; ++ti)
#pragma unroll
        for (int tj = 0; tj < 4; ++tj) acc[ti][tj] = zero8f();
    float ks_loc = 0.f;

    v8h   ka[2];
    float4 vb[4];

    // staging loader: clamped address + explicit zeroing (K axis!)
    auto load_regs = [&](int it) {
        int node = nStart + it * 32 + nl;
        bool valid = node < nEnd;
        int nc = valid ? node : (nEnd - 1);
#pragma unroll
        for (int j = 0; j < 2; ++j) {
            v8h v = *(const v8h*)(kf + (size_t)nc * P + mBase + seg * 16 + j * 8);
#pragma unroll
            for (int e = 0; e < 8; ++e) ka[j][e] = valid ? v[e] : (_Float16)0;
        }
#pragma unroll
        for (int j = 0; j < 4; ++j) {
            float4 v = *(const float4*)(src + (size_t)nc * D + dBase + seg * 16 + j * 4);
            vb[j].x = valid ? v.x : 0.f; vb[j].y = valid ? v.y : 0.f;
            vb[j].z = valid ? v.z : 0.f; vb[j].w = valid ? v.w : 0.f;
        }
    };
    auto store_regs = [&](int buf) {
#pragma unroll
        for (int j = 0; j < 2; ++j) {
            int chan = seg * 16 + j * 8;
#pragma unroll
            for (int e = 0; e < 8; ++e) lA[buf][(chan + e) * LDSK + nl] = ka[j][e];
        }
#pragma unroll
        for (int j = 0; j < 4; ++j) {
            int col = seg * 16 + j * 4;
            lB[buf][(col + 0) * LDSK + nl] = (_Float16)vb[j].x;
            lB[buf][(col + 1) * LDSK + nl] = (_Float16)vb[j].y;
            lB[buf][(col + 2) * LDSK + nl] = (_Float16)vb[j].z;
            lB[buf][(col + 3) * LDSK + nl] = (_Float16)vb[j].w;
        }
    };

    load_regs(0);
    store_regs(0);

#pragma unroll 1
    for (int it = 0; it < nIter; ++it) {
        __syncthreads();
        const int cur = it & 1;
        const bool more = (it + 1) < nIter;
        if (more) load_regs(it + 1);

        // fold ksum while K tile is resident (only one y-slice counts it)
        if (blockIdx.y == 0 && tid < 128) {
#pragma unroll
            for (int l = 0; l < 32; ++l) ks_loc += (float)lA[cur][tid * LDSK + l];
        }

        v16h af[2], bf[4];
#pragma unroll
        for (int ti = 0; ti < 2; ++ti) af[ti] = load_fragA(lA[cur], wm + ti * 16);
#pragma unroll
        for (int tj = 0; tj < 4; ++tj) bf[tj] = load_fragB(lB[cur], wn + tj * 16);
#pragma unroll
        for (int ti = 0; ti < 2; ++ti)
#pragma unroll
            for (int tj = 0; tj < 4; ++tj)
                acc[ti][tj] = wmma32(af[ti], bf[tj], acc[ti][tj]);

        if (more) store_regs(cur ^ 1);
    }

    const int cn  = lane & 15;
    const int cm0 = (lane >> 4) * 8;
#pragma unroll
    for (int ti = 0; ti < 2; ++ti)
#pragma unroll
        for (int tj = 0; tj < 4; ++tj)
#pragma unroll
            for (int i = 0; i < 8; ++i) {
                int mG = mBase + wm + ti * 16 + cm0 + i;
                int dG = dBase + wn + tj * 16 + cn;
                atomicAdd(&kv[(size_t)mG * D + dG], acc[ti][tj][i]);
            }
    if (blockIdx.y == 0 && tid < 128) atomicAdd(&ksum[mBase + tid], ks_loc);
}

// ---------------- kernel 3: output --------------------------------------
// out[n,d] = 0.25 * ( sum_h num_h*s/denom_h + N*v[n,d]*sum_h 1/denom_h )
// denom_h  = (Q[n,h]·ksum[h])*s + 1,  s = 1/(||Q|| ||K||)

__global__ void __launch_bounds__(256) out_kernel(
    const float* __restrict__ src, const _Float16* __restrict__ qf,
    const float* __restrict__ kv, const float* __restrict__ ksum,
    const float* __restrict__ ssq, float* __restrict__ out)
{
    __shared__ _Float16 lA[2][128 * LDSK];  // Q tile [node 128][k 32]
    __shared__ _Float16 lB[2][128 * LDSK];  // kv^T tile [d 128][k 32]
    __shared__ float lksum[P];
    __shared__ float lnorm[128 * NH];

    const int tid  = threadIdx.x;
    const int lane = tid & 31;
    const int wid  = tid >> 5;
    const int wm   = (wid & 3) * 32;
    const int wn   = (wid >> 2) * 64;
    const int mBase = blockIdx.x * 128;
    const int dBase = blockIdx.y * 128;
    const float inv_qk = rsqrtf(ssq[0] * ssq[1]);

    *(float4*)(lksum + tid * 4) = *(const float4*)(ksum + tid * 4);
    __syncthreads();

    // normalizer dots (clamped rows; invalid rows produce garbage that is
    // only consumed by guarded stores)
#pragma unroll
    for (int i = 0; i < 2; ++i) {
        int idx  = tid + i * 256;           // 128 nodes x 4 heads
        int nloc = idx >> 2, h = idx & 3;
        int nG   = mBase + nloc; if (nG >= N_NODES) nG = N_NODES - 1;
        const _Float16* qp = qf + (size_t)nG * P + h * D;
        const float*    kp = lksum + h * D;
        float dot = 0.f;
#pragma unroll
        for (int k8 = 0; k8 < D; k8 += 8) {
            v8h qv = *(const v8h*)(qp + k8);
#pragma unroll
            for (int e = 0; e < 8; ++e) dot += (float)qv[e] * kp[k8 + e];
        }
        lnorm[nloc * NH + h] = dot;
    }
    __syncthreads();

    // per-thread staging geometry
    int aRow[2], aSg[2], aNode[2];
#pragma unroll
    for (int i = 0; i < 2; ++i) {
        int v8id = tid + i * 256;
        aRow[i] = v8id >> 2; aSg[i] = (v8id & 3) * 8;
        int nG = mBase + aRow[i]; if (nG >= N_NODES) nG = N_NODES - 1;
        aNode[i] = nG;
    }

    v8f outAcc[2][4];
    float sAcc[2][8];
#pragma unroll
    for (int ti = 0; ti < 2; ++ti) {
#pragma unroll
        for (int tj = 0; tj < 4; ++tj) outAcc[ti][tj] = zero8f();
#pragma unroll
        for (int i = 0; i < 8; ++i) sAcc[ti][i] = 0.f;
    }

    const int cm0 = (lane >> 4) * 8;
    const int cn  = lane & 15;

#pragma unroll 1
    for (int h = 0; h < NH; ++h) {
        v8f accH[2][4];
#pragma unroll
        for (int ti = 0; ti < 2; ++ti)
#pragma unroll
            for (int tj = 0; tj < 4; ++tj) accH[ti][tj] = zero8f();

        float4 kb[4];

        // staging helpers
        auto stage_A = [&](int kc, int buf) {
#pragma unroll
            for (int i = 0; i < 2; ++i) {
                const _Float16* gp = qf + (size_t)aNode[i] * P + h * D + kc * 32 + aSg[i];
                _Float16* lp = &lA[buf][aRow[i] * LDSK + aSg[i]];
#ifdef HAVE_ASYNC_LDS
                async_ld_b128(gp, lp);     // ASYNCcnt-tracked direct global->LDS
#else
                *(v8h*)lp = *(const v8h*)gp;
#endif
            }
        };
        auto load_B = [&](int kc) {
#pragma unroll
            for (int i = 0; i < 4; ++i) {
                int f4 = tid + i * 256;
                int kl = f4 >> 5, d4 = (f4 & 31) * 4;
                kb[i] = *(const float4*)(kv + (size_t)(h * D + kc * 32 + kl) * D + dBase + d4);
            }
        };
        auto store_B = [&](int buf) {
#pragma unroll
            for (int i = 0; i < 4; ++i) {
                int f4 = tid + i * 256;
                int d4 = ((f4 & 31) * 4), kl = f4 >> 5;
                lB[buf][(d4 + 0) * LDSK + kl] = (_Float16)kb[i].x;
                lB[buf][(d4 + 1) * LDSK + kl] = (_Float16)kb[i].y;
                lB[buf][(d4 + 2) * LDSK + kl] = (_Float16)kb[i].z;
                lB[buf][(d4 + 3) * LDSK + kl] = (_Float16)kb[i].w;
            }
        };

        // prologue: chunk 0
        stage_A(0, 0);
        load_B(0);
        store_B(0);

#pragma unroll 1
        for (int kc = 0; kc < D / 32; ++kc) {
#ifdef HAVE_ASYNC_LDS
            wait_async0();                 // our async stores into LDS are done
#endif
            __syncthreads();
            const int cur = kc & 1;
            const bool more = (kc + 1) < D / 32;
            if (more) { stage_A(kc + 1, cur ^ 1); load_B(kc + 1); }

            v16h af[2], bf[4];
#pragma unroll
            for (int ti = 0; ti < 2; ++ti) af[ti] = load_fragA(lA[cur], wm + ti * 16);
#pragma unroll
            for (int tj = 0; tj < 4; ++tj) bf[tj] = load_fragB(lB[cur], wn + tj * 16);
#pragma unroll
            for (int ti = 0; ti < 2; ++ti)
#pragma unroll
                for (int tj = 0; tj < 4; ++tj)
                    accH[ti][tj] = wmma32(af[ti], bf[tj], accH[ti][tj]);

            if (more) store_B(cur ^ 1);
        }

        // fold this head
#pragma unroll
        for (int ti = 0; ti < 2; ++ti)
#pragma unroll
            for (int i = 0; i < 8; ++i) {
                int nloc = wm + ti * 16 + cm0 + i;
                float r = 1.0f / (lnorm[nloc * NH + h] * inv_qk + 1.0f);
                sAcc[ti][i] += r;
                float f = inv_qk * r;
#pragma unroll
                for (int tj = 0; tj < 4; ++tj)
                    outAcc[ti][tj][i] += accH[ti][tj][i] * f;
            }
        __syncthreads();   // head boundary: buffers about to be restaged
    }

    // final: out = 0.25 * (outAcc + N * v * sAcc)
#pragma unroll
    for (int ti = 0; ti < 2; ++ti)
#pragma unroll
        for (int tj = 0; tj < 4; ++tj)
#pragma unroll
            for (int i = 0; i < 8; ++i) {
                int nG = mBase + wm + ti * 16 + cm0 + i;
                int dG = dBase + wn + tj * 16 + cn;
                if (nG < N_NODES) {
                    float v = src[(size_t)nG * D + dG];
                    out[(size_t)nG * D + dG] =
                        0.25f * (outAcc[ti][tj][i] + (float)N_NODES * v * sAcc[ti][i]);
                }
            }
}

// ---------------- launch ----------------

extern "C" void kernel_launch(void* const* d_in, const int* in_sizes, int n_in,
                              void* d_out, int out_size, void* d_ws, size_t ws_size,
                              hipStream_t stream) {
    const float* query  = (const float*)d_in[0];
    const float* source = (const float*)d_in[1];
    const float* wq     = (const float*)d_in[2];
    const float* bq     = (const float*)d_in[3];
    const float* wk     = (const float*)d_in[4];
    const float* bk     = (const float*)d_in[5];

    float* wsf  = (float*)d_ws;
    float* ssq  = wsf + SSQ_F;
    float* ksum = wsf + KSUM_F;
    float* kv   = wsf + KV_F;
    _Float16* qf = (_Float16*)((char*)d_ws + QOFF_B);
    _Float16* kf = (_Float16*)((char*)d_ws + KOFF_B);
    float* out = (float*)d_out;

    const int mTiles = (N_NODES + 127) / 128;   // 391

    zero_kernel<<<(ZERO_TOTAL_F / 4 + 255) / 256, 256, 0, stream>>>(wsf, ZERO_TOTAL_F);
    proj_kernel<<<dim3(mTiles, P / 128, 2), 256, 0, stream>>>(
        query, source, wq, bq, wk, bk, ssq, qf, kf);
    kv_kernel<<<dim3(P / 128, D / 128, 32), 256, 0, stream>>>(source, kf, kv, ksum);
    out_kernel<<<dim3(mTiles, D / 128, 1), 256, 0, stream>>>(
        source, qf, kv, ksum, ssq, out);
}
